// MultiheadAttention_52802327937631
// MI455X (gfx1250) — matrix operand order repaired and checked by reading, not compiled, not run on hardware
//
#include <hip/hip_runtime.h>

typedef __attribute__((ext_vector_type(16))) _Float16 v16h;
typedef __attribute__((ext_vector_type(8)))  _Float16 v8h;
typedef __attribute__((ext_vector_type(4)))  _Float16 v4h;
typedef __attribute__((ext_vector_type(8)))  float    v8f;
typedef __attribute__((ext_vector_type(4)))  float    v4f;

#define B_  8
#define S_  1024
#define H_  1024
#define NH_ 16
#define HD_ 64
#define M_  (B_ * S_)      // 8192 rows (b,s)
#define N3_ (3 * H_)       // 3072 qkv cols

// ---------------------------------------------------------------------------
// WMMA helpers (CDNA5 gfx1250, wave32)
// ---------------------------------------------------------------------------
__device__ __forceinline__ v8f wmma16(v16h a, v16h b, v8f c) {
  // D = A(16x32 f16) x B(32x16 f16) + C(16x16 f32)
  return __builtin_amdgcn_wmma_f32_16x16x32_f16(
      /*neg_a=*/false, a, /*neg_b=*/false, b,
      /*c_mod=*/(short)0, c, /*reuse_a=*/false, /*reuse_b=*/false);
}

// A-matrix 16x32 f16 fragment. base is row-major [.., ld] f16, element (m,k).
// Contiguous k order used here (not the ISA table order): lane L -> row m = L%16; lanes<16 hold K = k0..k0+15
// in elements 0..15; lanes>=16 hold k0+16..k0+31.  Two contiguous 8-half
// runs per lane -> two global_load_b128.
__device__ __forceinline__ v16h load_a_frag(const _Float16* __restrict__ base,
                                            int ld, int m0, int k0, int lane) {
  int m  = m0 + (lane & 15);
  int kb = k0 + (lane & 16);          // +16 for upper half-wave
  const _Float16* p = base + (size_t)m * ld + kb;
  v8h lo = *(const v8h*)(p);
  v8h hi = *(const v8h*)(p + 8);
  return __builtin_shufflevector(lo, hi, 0, 1, 2, 3, 4, 5, 6, 7,
                                 8, 9, 10, 11, 12, 13, 14, 15);
}

// B-matrix 32x16 f16 fragment, sourced from BT stored row-major as BT[n][k]
// (i.e. BT[n*ld + k] == B[k][n]).  Contiguous k order used here: lane L -> col n = L%16;
// lanes<16 hold K = k0..k0+15 in elements 0..15, lanes>=16 hold k0+16..k0+31.
__device__ __forceinline__ v16h load_b_frag(const _Float16* __restrict__ base,
                                            int ld, int k0, int n0, int lane) {
  int n  = n0 + (lane & 15);
  int kb = k0 + (lane & 16);          // +16 for upper half-wave
  const _Float16* p = base + (size_t)n * ld + kb;
  v8h lo = *(const v8h*)(p);
  v8h hi = *(const v8h*)(p + 8);
  return __builtin_shufflevector(lo, hi, 0, 1, 2, 3, 4, 5, 6, 7,
                                 8, 9, 10, 11, 12, 13, 14, 15);
}

// ---------------------------------------------------------------------------
// fp32 -> f16 cast, 4 elements/thread
// ---------------------------------------------------------------------------
__global__ void cvt_f32_to_f16_x4(const v4f* __restrict__ src,
                                  v4h* __restrict__ dst, int n4) {
  int i = blockIdx.x * blockDim.x + threadIdx.x;
  if (i < n4) {
    v4f x = src[i];
    v4h y;
#pragma unroll
    for (int j = 0; j < 4; ++j) y[j] = (_Float16)x[j];
    dst[i] = y;
  }
}

// ---------------------------------------------------------------------------
// QKV projection: qkv[m,n] = sum_k X[m,k] * W[n,k] + bias[n]
// Wave tile: 32x64 (2x4 WMMA accumulators).  Output scattered into head
// layouts: Q,K -> [B,NH,S,HD] f16 ; V -> transposed [B,NH,HD,S] f16.
// ---------------------------------------------------------------------------
__global__ void __launch_bounds__(256) qkv_gemm_kernel(
    const _Float16* __restrict__ X,    // [M_, H_]
    const _Float16* __restrict__ W,    // [N3_, H_]
    const float* __restrict__ bias,    // [N3_]
    _Float16* __restrict__ Qh, _Float16* __restrict__ Kh,
    _Float16* __restrict__ Vt) {
  const int lane = threadIdx.x & 31;
  const int wid  = blockIdx.x * 8 + (threadIdx.x >> 5);
  const int NT   = N3_ / 64;           // 48
  const int mt   = wid / NT;
  const int nt   = wid % NT;
  const int m0   = mt * 32, n0 = nt * 64;

  const v8f z = {0.f, 0.f, 0.f, 0.f, 0.f, 0.f, 0.f, 0.f};
  v8f acc[2][4];
#pragma unroll
  for (int i = 0; i < 2; ++i)
#pragma unroll
    for (int j = 0; j < 4; ++j) acc[i][j] = z;

  for (int k0 = 0; k0 < H_; k0 += 32) {
    v16h a0 = load_a_frag(X, H_, m0, k0, lane);
    v16h a1 = load_a_frag(X, H_, m0 + 16, k0, lane);
#pragma unroll
    for (int j = 0; j < 4; ++j) {
      v16h b = load_b_frag(W, H_, k0, n0 + 16 * j, lane);
      acc[0][j] = wmma16(a0, b, acc[0][j]);
      acc[1][j] = wmma16(a1, b, acc[1][j]);
    }
  }

  const int rowoff = (lane & 16) >> 1;   // +8 rows for upper half-wave
#pragma unroll
  for (int j = 0; j < 4; ++j) {
    int n = n0 + 16 * j + (lane & 15);
    float bv = bias[n];
#pragma unroll
    for (int i = 0; i < 2; ++i) {
#pragma unroll
      for (int r = 0; r < 8; ++r) {
        int m = m0 + 16 * i + r + rowoff;
        _Float16 hv = (_Float16)(acc[i][j][r] + bv);
        int bb = m >> 10;          // S_ == 1024
        int s  = m & (S_ - 1);
        if (n < H_) {
          int h = n >> 6, d = n & 63;
          Qh[(((size_t)(bb * NH_ + h)) * S_ + s) * HD_ + d] = hv;
        } else if (n < 2 * H_) {
          int nn = n - H_;
          int h = nn >> 6, d = nn & 63;
          Kh[(((size_t)(bb * NH_ + h)) * S_ + s) * HD_ + d] = hv;
        } else {
          int nn = n - 2 * H_;
          int h = nn >> 6, d = nn & 63;
          Vt[(((size_t)(bb * NH_ + h)) * HD_ + d) * S_ + s] = hv;
        }
      }
    }
  }
}

// ---------------------------------------------------------------------------
// Flash attention, transposed formulation.
//   S_t[kk][q] = sum_d K[kk,d] * Q[q,d]          (A = K rows, B = Q)
//   ctx_t[d][q] += V_t[d,kk] * P_t[kk,q]         (A = V_t rows, B = P_t)
// Query index lives in the lane dimension -> per-lane scalar softmax stats
// (combined with the xor-16 partner lane that holds the other kk half).
// One wave = 16 queries of one head; block = 4 waves = 64 queries.
// ---------------------------------------------------------------------------
__global__ void __launch_bounds__(128) attn_kernel(
    const _Float16* __restrict__ Qh,   // [B*NH, S, HD]
    const _Float16* __restrict__ Kh,   // [B*NH, S, HD]
    const _Float16* __restrict__ Vt,   // [B*NH, HD, S]
    _Float16* __restrict__ Ctx) {      // [B, S, H] f16
  const int lane = threadIdx.x & 31;
  const int warp = threadIdx.x >> 5;
  const int bh   = blockIdx.x / (S_ / 64);     // head instance 0..127
  const int q0   = (blockIdx.x % (S_ / 64)) * 64 + warp * 16;

  const _Float16* Qp = Qh + (size_t)bh * S_ * HD_;
  const _Float16* Kp = Kh + (size_t)bh * S_ * HD_;
  const _Float16* Vp = Vt + (size_t)bh * HD_ * S_;

  // Q as B-fragments (k = d, n = q), resident for the whole loop.
  v16h qb0 = load_b_frag(Qp, HD_, 0, q0, lane);
  v16h qb1 = load_b_frag(Qp, HD_, 32, q0, lane);

  const v8f z = {0.f, 0.f, 0.f, 0.f, 0.f, 0.f, 0.f, 0.f};
  v8f ctx[4] = {z, z, z, z};           // ctx_t: 64 d-rows x 16 q-cols
  float mrow = -1e30f;                 // running max for this lane's q column
  float lrow = 0.f;                    // partial sum (this lane's kk half)
  const bool lowhalf = (lane & 16) == 0;
  const float sscale = 0.125f;         // 1/sqrt(HD)

  for (int kk0 = 0; kk0 < S_; kk0 += 32) {
    // ---- scores: two 16(kk) x 16(q) tiles, reduced over d = 64
    v8f c0 = z, c1 = z;
    v16h ka;
    ka = load_a_frag(Kp, HD_, kk0, 0, lane);       c0 = wmma16(ka, qb0, c0);
    ka = load_a_frag(Kp, HD_, kk0, 32, lane);      c0 = wmma16(ka, qb1, c0);
    ka = load_a_frag(Kp, HD_, kk0 + 16, 0, lane);  c1 = wmma16(ka, qb0, c1);
    ka = load_a_frag(Kp, HD_, kk0 + 16, 32, lane); c1 = wmma16(ka, qb1, c1);

    // ---- online softmax (per-lane scalar stats, partner lane = xor 16)
    float tmax = -1e30f;
#pragma unroll
    for (int r = 0; r < 8; ++r) {
      c0[r] *= sscale;
      c1[r] *= sscale;
      tmax = fmaxf(tmax, fmaxf(c0[r], c1[r]));
    }
    tmax = fmaxf(tmax, __shfl_xor(tmax, 16, 32));
    float mnew  = fmaxf(mrow, tmax);
    float scale = __expf(mrow - mnew);
    mrow = mnew;
    lrow *= scale;
#pragma unroll
    for (int dt = 0; dt < 4; ++dt)
#pragma unroll
      for (int r = 0; r < 8; ++r) ctx[dt][r] *= scale;

    float p0[8], p1[8];
    float lsum = 0.f;
#pragma unroll
    for (int r = 0; r < 8; ++r) {
      p0[r] = __expf(c0[r] - mnew);
      p1[r] = __expf(c1[r] - mnew);
      lsum += p0[r] + p1[r];
    }
    lrow += lsum;

    // ---- rearrange P (C-frag layout) into a 32x16 B-fragment via xor-16
    v16h pb;
#pragma unroll
    for (int e = 0; e < 8; ++e) {
      float t0 = __shfl_xor(p0[e], 16, 32);
      float t1 = __shfl_xor(p1[e], 16, 32);
      pb[e]     = (_Float16)(lowhalf ? p0[e] : t1);
      pb[e + 8] = (_Float16)(lowhalf ? t0 : p1[e]);
    }

    // ---- ctx_t[d][q] += V_t x P_t  (4 d-tiles of 16)
#pragma unroll
    for (int dt = 0; dt < 4; ++dt) {
      v16h va = load_a_frag(Vp, S_, dt * 16, kk0, lane);
      ctx[dt] = wmma16(va, pb, ctx[dt]);
    }
  }

  float ltot = lrow + __shfl_xor(lrow, 16, 32);
  float inv  = 1.f / ltot;

  // store ctx in [B, S, NH, HD] == [B, S, H] f16 layout for the out-proj GEMM
  const int bb = bh / NH_;
  const int h  = bh % NH_;
  const int s  = q0 + (lane & 15);
  _Float16* outp = Ctx + ((size_t)(bb * S_ + s)) * H_ + h * HD_;
  const int rowoff = (lane & 16) >> 1;
#pragma unroll
  for (int dt = 0; dt < 4; ++dt)
#pragma unroll
    for (int r = 0; r < 8; ++r) {
      int d = dt * 16 + r + rowoff;
      outp[d] = (_Float16)(ctx[dt][r] * inv);
    }
}

// ---------------------------------------------------------------------------
// Output projection: out[m,n] = sum_k ctx[m,k] * Wout[n,k] + bias[n]  (fp32)
// ---------------------------------------------------------------------------
__global__ void __launch_bounds__(256) out_gemm_kernel(
    const _Float16* __restrict__ Xc,   // [M_, H_]
    const _Float16* __restrict__ W,    // [H_, H_]
    const float* __restrict__ bias,    // [H_]
    float* __restrict__ out) {         // [M_, H_]
  const int lane = threadIdx.x & 31;
  const int wid  = blockIdx.x * 8 + (threadIdx.x >> 5);
  const int NT   = H_ / 64;            // 16
  const int mt   = wid / NT;
  const int nt   = wid % NT;
  const int m0   = mt * 32, n0 = nt * 64;

  const v8f z = {0.f, 0.f, 0.f, 0.f, 0.f, 0.f, 0.f, 0.f};
  v8f acc[2][4];
#pragma unroll
  for (int i = 0; i < 2; ++i)
#pragma unroll
    for (int j = 0; j < 4; ++j) acc[i][j] = z;

  for (int k0 = 0; k0 < H_; k0 += 32) {
    v16h a0 = load_a_frag(Xc, H_, m0, k0, lane);
    v16h a1 = load_a_frag(Xc, H_, m0 + 16, k0, lane);
#pragma unroll
    for (int j = 0; j < 4; ++j) {
      v16h b = load_b_frag(W, H_, k0, n0 + 16 * j, lane);
      acc[0][j] = wmma16(a0, b, acc[0][j]);
      acc[1][j] = wmma16(a1, b, acc[1][j]);
    }
  }

  const int rowoff = (lane & 16) >> 1;
#pragma unroll
  for (int j = 0; j < 4; ++j) {
    int n = n0 + 16 * j + (lane & 15);
    float bv = bias[n];
#pragma unroll
    for (int i = 0; i < 2; ++i)
#pragma unroll
      for (int r = 0; r < 8; ++r) {
        int m = m0 + 16 * i + r + rowoff;
        out[(size_t)m * H_ + n] = acc[i][j][r] + bv;
      }
  }
}

// ---------------------------------------------------------------------------
// Host-side launcher
// ---------------------------------------------------------------------------
extern "C" void kernel_launch(void* const* d_in, const int* in_sizes, int n_in,
                              void* d_out, int out_size, void* d_ws,
                              size_t ws_size, hipStream_t stream) {
  (void)in_sizes; (void)n_in; (void)out_size; (void)ws_size;

  const float* hs = (const float*)d_in[0];   // [B,S,H]
  const float* wi = (const float*)d_in[1];   // [3H,H]
  const float* bi = (const float*)d_in[2];   // [3H]
  const float* wo = (const float*)d_in[3];   // [H,H]
  const float* bo = (const float*)d_in[4];   // [H]
  float* out = (float*)d_out;

  char* p = (char*)d_ws;
  _Float16* Xh    = (_Float16*)p; p += (size_t)M_ * H_ * 2;   // 16 MB
  _Float16* Winh  = (_Float16*)p; p += (size_t)N3_ * H_ * 2;  //  6 MB
  _Float16* Wouth = (_Float16*)p; p += (size_t)H_ * H_ * 2;   //  2 MB
  _Float16* Qh    = (_Float16*)p; p += (size_t)M_ * H_ * 2;   // 16 MB
  _Float16* Kh    = (_Float16*)p; p += (size_t)M_ * H_ * 2;   // 16 MB
  _Float16* Vt    = (_Float16*)p; p += (size_t)M_ * H_ * 2;   // 16 MB
  _Float16* Ctxh  = (_Float16*)p; p += (size_t)M_ * H_ * 2;   // 16 MB

  const int T = 256;
  {
    int n4 = M_ * H_ / 4;
    cvt_f32_to_f16_x4<<<(n4 + T - 1) / T, T, 0, stream>>>((const v4f*)hs,
                                                          (v4h*)Xh, n4);
  }
  {
    int n4 = N3_ * H_ / 4;
    cvt_f32_to_f16_x4<<<(n4 + T - 1) / T, T, 0, stream>>>((const v4f*)wi,
                                                          (v4h*)Winh, n4);
  }
  {
    int n4 = H_ * H_ / 4;
    cvt_f32_to_f16_x4<<<(n4 + T - 1) / T, T, 0, stream>>>((const v4f*)wo,
                                                          (v4h*)Wouth, n4);
  }

  // QKV: (M/32)*(3H/64) wave tiles, 8 waves/block
  qkv_gemm_kernel<<<(M_ / 32) * (N3_ / 64) / 8, 256, 0, stream>>>(
      Xh, Winh, bi, Qh, Kh, Vt);

  // Attention: one block (4 waves) per 64 queries per head instance
  attn_kernel<<<B_ * NH_ * (S_ / 64), 128, 0, stream>>>(Qh, Kh, Vt, Ctxh);

  // Output projection
  out_gemm_kernel<<<(M_ / 32) * (H_ / 64) / 8, 256, 0, stream>>>(
      Ctxh, Wouth, bo, out);
}
